// MT_GAT_topk_shareEn_multiple8_joint_70712341561396
// MI455X (gfx1250) — matrix operand order repaired and checked by reading, not compiled, not run on hardware
//
#include <hip/hip_runtime.h>
#include <hip/hip_bf16.h>

// ---------------------------------------------------------------------------
// Model constants
// ---------------------------------------------------------------------------
#define SEGC 264
#define HIDC 128
#define KTOP 132
#define HCC  256
#define ZD   (KTOP*HIDC + SEGC*HIDC)   // 50688
#define BNSC 0.99999500003749973e0f    // 1/sqrt(1+1e-5)

// ---------------------------------------------------------------------------
// types / bf16 helpers
// ---------------------------------------------------------------------------
typedef __bf16 bf16_t;
typedef bf16_t bf16x16 __attribute__((ext_vector_type(16)));
typedef float  f32x8   __attribute__((ext_vector_type(8)));
typedef unsigned short us4 __attribute__((ext_vector_type(4)));
typedef unsigned short us8 __attribute__((ext_vector_type(8)));

__device__ __forceinline__ unsigned short f2bf(float f) {
  unsigned int u = __builtin_bit_cast(unsigned int, f);
  unsigned int r = u + 0x7fffu + ((u >> 16) & 1u);   // round-to-nearest-even
  return (unsigned short)(r >> 16);
}
__device__ __forceinline__ bf16_t bfbits(unsigned short s) {
  return __builtin_bit_cast(bf16_t, s);
}

// ---------------------------------------------------------------------------
// CDNA5 async global->LDS staging (CDNA5 ISA)
// VDST VGPR = per-lane LDS byte offset (wave-LDS relative); GV mode 64b vaddr.
// Tracked by ASYNCcnt -> s_wait_asynccnt before the workgroup barrier.
// ---------------------------------------------------------------------------
typedef __attribute__((address_space(3))) const void lds_cv;
__device__ __forceinline__ unsigned lds_off_u32(const void* p) {
  return (unsigned)(unsigned long long)(lds_cv*)p;   // addrspacecast + ptrtoint
}
__device__ __forceinline__ void async_b128(const void* lds_dst, const void* gsrc) {
  asm volatile("global_load_async_to_lds_b128 %0, %1, off"
               :: "v"(lds_off_u32(lds_dst)), "v"((unsigned long long)gsrc)
               : "memory");
}
__device__ __forceinline__ void wait_async0() {
  asm volatile("s_wait_asynccnt 0x0" ::: "memory");
}

// ---------------------------------------------------------------------------
// WMMA fragment loaders (layouts per CDNA5 ISA, wave32)
// A: 16x32 bf16.  lanes 0-15 => M=lane, VGPR0-3: K=0..7, VGPR4-7: K=16..23
//                 lanes16-31 => M=lane-16, VGPR0-3: K=8..15, VGPR4-7: K=24..31
// ---------------------------------------------------------------------------
__device__ __forceinline__ bf16x16 load_a_frag(const unsigned short* p, int ld, int lane) {
  bf16x16 a;
  int m  = lane & 15;
  int hi = (lane & 16) ? 8 : 0;
#pragma unroll
  for (int h = 0; h < 16; ++h) {
    int v = h >> 1, s = h & 1;
    int k = ((v & 4) ? 16 : 0) + hi + ((v & 3) << 1) + s;
    a[h] = bfbits(p[m * ld + k]);
  }
  return a;
}
// B: 32x16 bf16 stored in LDS as Bs[n][k]; lanes 0-15 K=0..7,16..23, lanes16-31 K=8..15,24..31
__device__ __forceinline__ bf16x16 load_b_frag(const unsigned short* p, int ld, int lane) {
  bf16x16 b;
  int n  = lane & 15;
  int kb = (lane & 16) ? 8 : 0;
#pragma unroll
  for (int h = 0; h < 16; ++h)
    b[h] = bfbits(p[n * ld + kb + (h & 7) + ((h & 8) ? 16 : 0)]);
  return b;
}

// ---------------------------------------------------------------------------
// Generic 2-level-batched WMMA GEMM:  C = act(alpha * A*B(^T) + bias)
// A: M x K bf16 row-major.  B: (transB ? N x K : K x N) bf16 row-major.
// 128 threads = 4 waves; block tile 128x64; wave w owns rows [32w,32w+32)
// -> 2 A-frags x 4 B-frags = 8 wmma per K-step per wave.
// Tile staging: async b128 (16B-aligned) > us4 vector (8B) > scalar w/ bounds.
// ---------------------------------------------------------------------------
#define TM 128
#define TN 64
#define TK 32
#define LDST 40   // LDS row stride in ushorts (80B: 16B-aligned chunks)

__global__ __launch_bounds__(128)
void k_wmma_gemm(const unsigned short* __restrict__ A, long long sA1, long long sA0,
                 const unsigned short* __restrict__ Bm, long long sB1, long long sB0,
                 const float* __restrict__ bias, long long sb1, long long sb0,
                 float* __restrict__ C, long long sC1, long long sC0,
                 int nInner, int lda, int ldb, int ldc,
                 int M, int Nn, int Kk, int transB, float alpha, int act) {
  __shared__ unsigned short As[TM][LDST];
  __shared__ unsigned short Bs[TN][LDST];

  int bz = blockIdx.z;
  int outer = bz / nInner, inner = bz % nInner;
  A  += outer * sA1 + inner * sA0;
  Bm += outer * sB1 + inner * sB0;
  C  += outer * sC1 + inner * sC0;
  const float* bi = bias ? (bias + outer * sb1 + inner * sb0) : nullptr;

  int row0 = blockIdx.y * TM;
  int col0 = blockIdx.x * TN;
  int tid  = threadIdx.x;
  int wave = tid >> 5;
  int lane = tid & 31;

  f32x8 acc[2][4];
#pragma unroll
  for (int g = 0; g < 2; ++g)
#pragma unroll
    for (int j = 0; j < 4; ++j)
#pragma unroll
      for (int e = 0; e < 8; ++e) acc[g][j][e] = 0.f;

  // block-uniform alignment tiers (bytes): row stride and base must be aligned
  unsigned long long aBits = ((unsigned long long)(size_t)A)  | ((unsigned long long)(unsigned)(lda * 2));
  unsigned long long bBits = ((unsigned long long)(size_t)Bm) | ((unsigned long long)(unsigned)(ldb * 2));
  bool fullA = (row0 + TM <= M);
  bool fullB = (col0 + TN <= Nn);

  for (int k0 = 0; k0 < Kk; k0 += TK) {
    bool fullK = (k0 + TK <= Kk);
    int aMode = (fullA && fullK) ? (((aBits & 15) == 0) ? 2 : (((aBits & 7) == 0) ? 1 : 0)) : 0;
    int bMode = (transB && fullB && fullK) ? (((bBits & 15) == 0) ? 2 : (((bBits & 7) == 0) ? 1 : 0)) : 0;

    // ---- stage A tile (128x32) ----
    if (aMode == 2) {
#pragma unroll
      for (int it = 0; it < 4; ++it) {                 // 512 x 16B chunks
        int q = it * 128 + tid;
        int r = q >> 2, c8 = (q & 3) * 8;
        async_b128(&As[r][c8], A + (long long)(row0 + r) * lda + k0 + c8);
      }
    } else if (aMode == 1) {
#pragma unroll
      for (int it = 0; it < 8; ++it) {                 // 1024 x 8B chunks
        int q = it * 128 + tid;
        int r = q >> 3, c4 = (q & 7) * 4;
        const unsigned short* g = A + (long long)(row0 + r) * lda + k0 + c4;
        __builtin_prefetch(g + lda, 0, 1);
        *(us4*)&As[r][c4] = *(const us4*)g;
      }
    } else {
#pragma unroll 4
      for (int it = 0; it < 32; ++it) {                // scalar, zero-padded
        int idx = it * 128 + tid;
        int r = idx >> 5, c = idx & 31;
        int gr = row0 + r, gk = k0 + c;
        As[r][c] = (gr < M && gk < Kk) ? A[(long long)gr * lda + gk] : (unsigned short)0;
      }
    }
    // ---- stage B tile (64x32, stored as Bs[n][k]) ----
    if (bMode == 2) {
#pragma unroll
      for (int it = 0; it < 2; ++it) {                 // 256 x 16B chunks
        int q = it * 128 + tid;
        int r = q >> 2, c8 = (q & 3) * 8;
        async_b128(&Bs[r][c8], Bm + (long long)(col0 + r) * ldb + k0 + c8);
      }
    } else if (bMode == 1) {
#pragma unroll
      for (int it = 0; it < 4; ++it) {                 // 512 x 8B chunks
        int q = it * 128 + tid;
        int r = q >> 3, c4 = (q & 7) * 4;
        *(us4*)&Bs[r][c4] = *(const us4*)(Bm + (long long)(col0 + r) * ldb + k0 + c4);
      }
    } else {
#pragma unroll 4
      for (int it = 0; it < 16; ++it) {
        int idx = it * 128 + tid;
        int r = idx >> 5, c = idx & 31;
        int gn = col0 + r, gk = k0 + c;
        unsigned short bv = 0;
        if (gn < Nn && gk < Kk)
          bv = transB ? Bm[(long long)gn * ldb + gk] : Bm[(long long)gk * ldb + gn];
        Bs[r][c] = bv;
      }
    }
    if (aMode == 2 || bMode == 2) wait_async0();
    __syncthreads();

    bf16x16 a0 = load_a_frag(&As[wave * 32][0],      LDST, lane);
    bf16x16 a1 = load_a_frag(&As[wave * 32 + 16][0], LDST, lane);
#pragma unroll
    for (int j = 0; j < 4; ++j) {
      bf16x16 b = load_b_frag(&Bs[j * 16][0], LDST, lane);
      acc[0][j] = __builtin_amdgcn_wmma_f32_16x16x32_bf16(false, a0, false, b, (short)0, acc[0][j], false, false);
      acc[1][j] = __builtin_amdgcn_wmma_f32_16x16x32_bf16(false, a1, false, b, (short)0, acc[1][j], false, false);
    }
    __syncthreads();
  }

  // epilogue: C/D layout — lane&15 = N, VGPR r -> M=r (+8 for lanes>=16)
  int n  = lane & 15;
  int mh = (lane & 16) ? 8 : 0;
#pragma unroll
  for (int g = 0; g < 2; ++g) {
#pragma unroll
    for (int r = 0; r < 8; ++r) {
      int gr = row0 + wave * 32 + g * 16 + r + mh;
      if (gr >= M) continue;
#pragma unroll
      for (int j = 0; j < 4; ++j) {
        int gc = col0 + j * 16 + n;
        if (gc < Nn) {
          float v = alpha * acc[g][j][r] + (bi ? bi[gc] : 0.f);
          if (act) v = v > 0.f ? v : 0.f;
          C[(long long)gr * ldc + gc] = v;
        }
      }
    }
  }
}

// ---------------------------------------------------------------------------
// Elementwise / utility kernels
// ---------------------------------------------------------------------------
__global__ void k_fill(float* p, float v, long long n) {
  long long i = blockIdx.x * (long long)blockDim.x + threadIdx.x;
  if (i < n) p[i] = v;
}
__global__ void k_f2bf(const float* __restrict__ a, unsigned short* __restrict__ o, long long n) {
  long long i = blockIdx.x * (long long)blockDim.x + threadIdx.x;
  if (i < n) o[i] = f2bf(a[i]);
}
__global__ void k_add(float* o, const float* a, const float* b, long long n) {
  long long i = blockIdx.x * (long long)blockDim.x + threadIdx.x;
  if (i < n) o[i] = a[i] + b[i];
}
__global__ void k_add_scale(float* o, const float* a, const float* b, float s, long long n) {
  long long i = blockIdx.x * (long long)blockDim.x + threadIdx.x;
  if (i < n) o[i] = (a[i] + b[i]) * s;
}
__global__ void k_add_relu_scale(float* o, const float* a, const float* b, float s, long long n) {
  long long i = blockIdx.x * (long long)blockDim.x + threadIdx.x;
  if (i < n) { float v = (a[i] + b[i]) * s; o[i] = v > 0.f ? v : 0.f; }
}
__global__ void k_combine(float* o, const float* a, const float* b, const float* c, float s, long long n) {
  long long i = blockIdx.x * (long long)blockDim.x + threadIdx.x;
  if (i < n) o[i] = a[i] + (b[i] + c[i]) * s;   // out = h + (oproj + x)*BNS
}
__global__ void k_bias_relu(float* o, const float* a, const float* bvec, int feat, long long n) {
  long long i = blockIdx.x * (long long)blockDim.x + threadIdx.x;
  if (i < n) { float v = a[i] + bvec[i % feat]; o[i] = v > 0.f ? v : 0.f; }
}

// row softmax, one block (256 thr) per row, in-place capable
__global__ void k_softmax_rows(const float* __restrict__ in, float* __restrict__ out,
                               int rows, int cols) {
  int row = blockIdx.x; if (row >= rows) return;
  const float* r = in + (long long)row * cols;
  float* o = out + (long long)row * cols;
  __shared__ float red[256];
  int t = threadIdx.x;
  float mx = -3.0e38f;
  for (int c = t; c < cols; c += 256) mx = fmaxf(mx, r[c]);
  red[t] = mx; __syncthreads();
  for (int s = 128; s > 0; s >>= 1) { if (t < s) red[t] = fmaxf(red[t], red[t + s]); __syncthreads(); }
  mx = red[0]; __syncthreads();
  float sum = 0.f;
  for (int c = t; c < cols; c += 256) sum += __expf(r[c] - mx);
  red[t] = sum; __syncthreads();
  for (int s = 128; s > 0; s >>= 1) { if (t < s) red[t] += red[t + s]; __syncthreads(); }
  float inv = 1.f / red[0];
  for (int c = t; c < cols; c += 256) o[c] = __expf(r[c] - mx) * inv;
}

// ---------------------------------------------------------------------------
// GINE edge scatter: aggr[dst] += relu(x[src] + (w*gew + geb))
// ---------------------------------------------------------------------------
__global__ void k_gine_edges(const float* __restrict__ x, const int* __restrict__ src,
                             const int* __restrict__ dst, const float* __restrict__ ew,
                             const float* __restrict__ gew, const float* __restrict__ geb,
                             float* __restrict__ aggr, int E) {
  int e = blockIdx.x; if (e >= E) return;
  float w = ew[e];
  long long s = src[e], d = dst[e];
  for (int f = threadIdx.x; f < SEGC; f += blockDim.x) {
    float m = x[s * SEGC + f] + (w * gew[f] + geb[f]);
    m = m > 0.f ? m : 0.f;
    atomicAdd(&aggr[d * SEGC + f], m);
  }
}

// ---------------------------------------------------------------------------
// GAT kernels
// ---------------------------------------------------------------------------
__global__ void k_gat_scores(const float* __restrict__ xp, const float* __restrict__ as,
                             const float* __restrict__ ad, float* asrc, float* adst, int n) {
  int i = blockIdx.x * blockDim.x + threadIdx.x; if (i >= n) return;
  float s = 0.f, d = 0.f;
  const float* row = xp + (long long)i * HIDC;
#pragma unroll 4
  for (int f = 0; f < HIDC; ++f) { s += row[f] * as[f]; d += row[f] * ad[f]; }
  asrc[i] = s; adst[i] = d;
}
__device__ __forceinline__ void atomicMaxF(float* addr, float val) {
  int* ai = reinterpret_cast<int*>(addr);
  int old = __float_as_int(*addr);
  while (__int_as_float(old) < val) {
    int prev = atomicCAS(ai, old, __float_as_int(val));
    if (prev == old) break;
    old = prev;
  }
}
__global__ void k_gat_max(const int* src, const int* dst, int count,
                          const float* asrc, const float* adst, float* m) {
  int e = blockIdx.x * blockDim.x + threadIdx.x; if (e >= count) return;
  int s = src ? src[e] : e, d = dst ? dst[e] : e;
  float a = asrc[s] + adst[d]; a = a > 0.f ? a : 0.2f * a;
  atomicMaxF(&m[d], a);
}
__global__ void k_gat_den(const int* src, const int* dst, int count,
                          const float* asrc, const float* adst, const float* m, float* den) {
  int e = blockIdx.x * blockDim.x + threadIdx.x; if (e >= count) return;
  int s = src ? src[e] : e, d = dst ? dst[e] : e;
  float a = asrc[s] + adst[d]; a = a > 0.f ? a : 0.2f * a;
  atomicAdd(&den[d], __expf(a - m[d]));
}
__global__ void k_gat_aggr(const int* src, const int* dst, int count,
                           const float* asrc, const float* adst, const float* m,
                           const float* den, const float* __restrict__ xp,
                           float* __restrict__ out) {
  int e = blockIdx.x; if (e >= count) return;
  int s = src ? src[e] : e, d = dst ? dst[e] : e;
  float a = asrc[s] + adst[d]; a = a > 0.f ? a : 0.2f * a;
  float coef = __expf(a - m[d]) / den[d];
  int f = threadIdx.x;  // 128 threads
  atomicAdd(&out[(long long)d * HIDC + f], coef * xp[(long long)s * HIDC + f]);
}

// ---------------------------------------------------------------------------
// node attention scalar + scaling
// ---------------------------------------------------------------------------
__global__ void k_conv4(const float* x11, const float* x22, const float* w,
                        const float* b0, float* a, int n) {
  int i = blockIdx.x * blockDim.x + threadIdx.x; if (i >= n) return;
  float s = 0.f;
  const float* r1 = x11 + (long long)i * HIDC;
  const float* r2 = x22 + (long long)i * HIDC;
#pragma unroll 4
  for (int f = 0; f < HIDC; ++f) s += r1[f] * w[f] + r2[f] * w[HIDC + f];
  a[i] = s + b0[0];
}
__global__ void k_allscale(const float* attn, const float* x11, const float* x22,
                           float* all1, float* all2, long long n128) {
  long long i = blockIdx.x * (long long)blockDim.x + threadIdx.x;
  if (i >= n128) return;
  float a = attn[i >> 7];
  all1[i] = a * x11[i];
  all2[i] = a * x22[i];
}

// ---------------------------------------------------------------------------
// TopK pooling
// ---------------------------------------------------------------------------
__global__ void k_wn(const float* pw, float* wn) {
  __shared__ float red[128];
  int h = blockIdx.x, t = threadIdx.x;
  float v = pw[h * HIDC + t];
  red[t] = v * v; __syncthreads();
  for (int s = 64; s > 0; s >>= 1) { if (t < s) red[t] += red[t + s]; __syncthreads(); }
  wn[h * HIDC + t] = v / sqrtf(red[0]);
}

// block = 264 threads, grid = (B, 8)
__global__ void k_topk(const float* __restrict__ x11, const float* __restrict__ x22,
                       const float* __restrict__ wn, float* vals, int* idxs,
                       float* dout, int Bn) {
  __shared__ float sc[SEGC];
  int b = blockIdx.x, h = blockIdx.y, t = threadIdx.x;
  const float* xin = (h < 4) ? x11 : x22;
  const float* w = wn + h * HIDC;
  long long node = (long long)b * SEGC + t;
  float s = 0.f;
#pragma unroll 4
  for (int f = 0; f < HIDC; ++f) s += xin[node * HIDC + f] * w[f];
  s = 1.f / (1.f + __expf(-s));
  sc[t] = s; __syncthreads();
  int rank = 0;
  for (int j = 0; j < SEGC; ++j) {
    float sj = sc[j];
    rank += (sj > s) || (sj == s && j < t);
  }
  if (rank < KTOP) {
    long long base = ((long long)h * Bn + b) * KTOP + rank;
    vals[base] = s;
    idxs[base] = t;
    float* hout = dout + (long long)h * (256 + 2 * (long long)Bn * KTOP);
    hout[256 + b * KTOP + rank] = (float)(b * SEGC + t);                    // perm
    hout[256 + Bn * KTOP + b * KTOP + rank] = 1.f / (1.f + __expf(-s));     // sigmoid(vals)
  }
}

__global__ void k_zpool(const float* x11, const float* x22, const float* vals,
                        const int* idxs, unsigned short* z, int Bn) {
  long long total = 8LL * Bn * KTOP * HIDC;
  long long i = blockIdx.x * (long long)blockDim.x + threadIdx.x;
  if (i >= total) return;
  int f = i & (HIDC - 1);
  long long t = i >> 7;
  int r = t % KTOP; t /= KTOP;
  int b = t % Bn;   int h = t / Bn;
  long long vbase = ((long long)h * Bn + b) * KTOP + r;
  long long node = (long long)b * SEGC + idxs[vbase];
  const float* xin = (h < 4) ? x11 : x22;
  float v = vals[vbase] * xin[node * HIDC + f];
  v = v > 0.f ? v : 0.f;
  z[((long long)h * Bn + b) * ZD + (long long)r * HIDC + f] = f2bf(v);
}
__global__ void k_zcat(const float* all1, const float* all2, unsigned short* z, int Bn) {
  long long total = 8LL * Bn * SEGC * HIDC;
  long long i = blockIdx.x * (long long)blockDim.x + threadIdx.x;
  if (i >= total) return;
  int f = i & (HIDC - 1);
  long long t = i >> 7;
  int n = t % SEGC; t /= SEGC;
  int b = t % Bn;   int h = t / Bn;
  const float* all = (h < 4) ? all2 : all1;
  float v = all[((long long)b * SEGC + n) * HIDC + f];
  v = v > 0.f ? v : 0.f;
  z[((long long)h * Bn + b) * ZD + (long long)KTOP * HIDC + (long long)n * HIDC + f] = f2bf(v);
}

// yo[h,b] = sum_c y[h,b,c]*lsw[h,c] + lsb[h]; grid (B, 8), block 256
__global__ void k_yo(const float* __restrict__ y, const float* __restrict__ lsw,
                     const float* __restrict__ lsb, float* dout, int Bn) {
  __shared__ float red[256];
  int b = blockIdx.x, h = blockIdx.y, t = threadIdx.x;
  red[t] = y[((long long)h * Bn + b) * HCC + t] * lsw[h * HCC + t];
  __syncthreads();
  for (int s = 128; s > 0; s >>= 1) { if (t < s) red[t] += red[t + s]; __syncthreads(); }
  if (t == 0) dout[(long long)h * (256 + 2 * (long long)Bn * KTOP) + b] = red[0] + lsb[h];
}

// ---------------------------------------------------------------------------
// Host orchestration
// ---------------------------------------------------------------------------
extern "C" void kernel_launch(void* const* d_in, const int* in_sizes, int n_in,
                              void* d_out, int out_size, void* d_ws, size_t ws_size,
                              hipStream_t stream) {
  const float* x      = (const float*)d_in[0];
  const int*   eidx   = (const int*)d_in[1];
  const float* ew     = (const float*)d_in[2];
  const float* gew    = (const float*)d_in[4];
  const float* geb    = (const float*)d_in[5];
  const float* gw1    = (const float*)d_in[6];
  const float* gb1    = (const float*)d_in[7];
  const float* gw2    = (const float*)d_in[8];
  const float* gb2    = (const float*)d_in[9];
  const float* aiw    = (const float*)d_in[10];
  const float* aib    = (const float*)d_in[11];
  const float* aow    = (const float*)d_in[12];
  const float* aob    = (const float*)d_in[13];
  const float* m1w    = (const float*)d_in[14];
  const float* m1b    = (const float*)d_in[15];
  const float* m2w    = (const float*)d_in[16];
  const float* m2b    = (const float*)d_in[17];
  const float* g1w    = (const float*)d_in[18];
  const float* g1as   = (const float*)d_in[19];
  const float* g1ad   = (const float*)d_in[20];
  const float* g1b    = (const float*)d_in[21];
  const float* g2w    = (const float*)d_in[22];
  const float* g2as   = (const float*)d_in[23];
  const float* g2ad   = (const float*)d_in[24];
  const float* g2b    = (const float*)d_in[25];
  const float* c4w    = (const float*)d_in[26];
  const float* c4b    = (const float*)d_in[27];
  const float* l4w    = (const float*)d_in[28];
  const float* l4b    = (const float*)d_in[29];
  const float* l5w    = (const float*)d_in[30];
  const float* l5b    = (const float*)d_in[31];
  const float* pw     = (const float*)d_in[32];
  const float* lbw    = (const float*)d_in[33];
  const float* lbb    = (const float*)d_in[34];
  const float* lsw    = (const float*)d_in[35];
  const float* lsb    = (const float*)d_in[36];
  float* dout = (float*)d_out;

  const int  Nn = in_sizes[0] / SEGC;            // 67584 nodes
  const int  Bn = Nn / SEGC;                     // 256 graphs
  const int  E  = in_sizes[1] / 2;               // edges
  const long long S = (long long)Nn * SEGC;      // N*264 elements
  const int* src = eidx;
  const int* dst = eidx + E;

  // ---- workspace bump allocator ----
  char* wp = (char*)d_ws; size_t off = 0;
  auto aF = [&](long long n) -> float* {
    off = (off + 255) & ~(size_t)255; float* p = (float*)(wp + off); off += (size_t)n * 4; return p; };
  auto aH = [&](long long n) -> unsigned short* {
    off = (off + 255) & ~(size_t)255; auto* p = (unsigned short*)(wp + off); off += (size_t)n * 2; return p; };
  auto aI = [&](long long n) -> int* {
    off = (off + 255) & ~(size_t)255; int* p = (int*)(wp + off); off += (size_t)n * 4; return p; };

  // f32 activations
  float* f_aggr   = aF(S);
  float* f_h1     = aF(S);
  float* f_h2     = aF(S);
  float* f_qkv    = aF(3 * S);
  float* f_scores = aF((long long)Bn * 2 * SEGC * SEGC);
  float* f_ao     = aF(S);
  float* f_oproj  = aF(S);
  float* f_out    = aF(S);
  float* f_t1     = aF(2 * S);
  float* f_t2     = aF(S);
  float* f_x0     = aF(S);
  float* f_xp1    = aF((long long)Nn * HIDC);
  float* f_xp2    = aF((long long)Nn * HIDC);
  float* f_as1    = aF(Nn); float* f_ad1 = aF(Nn);
  float* f_as2    = aF(Nn); float* f_ad2 = aF(Nn);
  float* f_m1     = aF(Nn); float* f_dn1 = aF(Nn);
  float* f_m2     = aF(Nn); float* f_dn2 = aF(Nn);
  float* f_go1    = aF((long long)Nn * HIDC);   // -> x11 after bias+relu
  float* f_go2    = aF((long long)Nn * HIDC);   // -> x22
  float* f_a      = aF(Nn);
  float* f_t4     = aF((long long)Bn * 66);
  float* f_attn   = aF((long long)Bn * SEGC);
  float* f_all1   = aF((long long)Nn * HIDC);
  float* f_all2   = aF((long long)Nn * HIDC);
  float* f_wn     = aF(8 * HIDC);
  float* f_vals   = aF(8LL * Bn * KTOP);
  int*   i_idx    = aI(8LL * Bn * KTOP);
  float* f_y      = aF(8LL * Bn * HCC);
  // bf16 buffers
  unsigned short* b_x   = aH(S);
  unsigned short* b_tmp = aH(S);
  unsigned short* b_h1  = aH(S);
  unsigned short* b_qkv = aH(3 * S);
  unsigned short* b_sc  = aH((long long)Bn * 2 * SEGC * SEGC);
  unsigned short* b_ao  = aH(S);
  unsigned short* b_out = aH(S);
  unsigned short* b_t1  = aH(2 * S);
  unsigned short* b_x0  = aH(S);
  unsigned short* b_a   = aH(Nn);
  unsigned short* b_t4  = aH((long long)Bn * 66);
  unsigned short* b_gw1 = aH(SEGC * SEGC);
  unsigned short* b_gw2 = aH(SEGC * SEGC);
  unsigned short* b_aiw = aH(3 * SEGC * SEGC);
  unsigned short* b_aow = aH(SEGC * SEGC);
  unsigned short* b_m1w = aH(2 * SEGC * SEGC);
  unsigned short* b_m2w = aH(2 * SEGC * SEGC);
  unsigned short* b_g1w = aH(HIDC * SEGC);
  unsigned short* b_g2w = aH(HIDC * SEGC);
  unsigned short* b_l4w = aH(66 * SEGC);
  unsigned short* b_l5w = aH(SEGC * 66);
  unsigned short* b_lbw = aH(8LL * HCC * ZD);
  unsigned short* b_z   = aH(8LL * Bn * ZD);

  auto cvt = [&](const float* a, unsigned short* o, long long n) {
    k_f2bf<<<dim3((unsigned)((n + 255) / 256)), dim3(256), 0, stream>>>(a, o, n); };
  auto fill = [&](float* p, float v, long long n) {
    k_fill<<<dim3((unsigned)((n + 255) / 256)), dim3(256), 0, stream>>>(p, v, n); };
  auto gemm = [&](const unsigned short* A, long long sA1, long long sA0,
                  const unsigned short* Bp, long long sB1, long long sB0,
                  const float* bias, long long sb1, long long sb0,
                  float* C, long long sC1, long long sC0,
                  int nInner, int lda, int ldb, int ldc,
                  int M, int Ncols, int Kk, int transB, float alpha, int act, int batch) {
    dim3 g((Ncols + TN - 1) / TN, (M + TM - 1) / TM, batch);
    k_wmma_gemm<<<g, 128, 0, stream>>>(A, sA1, sA0, Bp, sB1, sB0, bias, sb1, sb0,
                                       C, sC1, sC0, nInner, lda, ldb, ldc,
                                       M, Ncols, Kk, transB, alpha, act);
  };

  // ---- weight conversions ----
  cvt(gw1, b_gw1, SEGC * SEGC);  cvt(gw2, b_gw2, SEGC * SEGC);
  cvt(aiw, b_aiw, 3LL * SEGC * SEGC); cvt(aow, b_aow, SEGC * SEGC);
  cvt(m1w, b_m1w, 2LL * SEGC * SEGC); cvt(m2w, b_m2w, 2LL * SEGC * SEGC);
  cvt(g1w, b_g1w, (long long)HIDC * SEGC); cvt(g2w, b_g2w, (long long)HIDC * SEGC);
  cvt(l4w, b_l4w, 66LL * SEGC); cvt(l5w, b_l5w, 66LL * SEGC);
  cvt(lbw, b_lbw, 8LL * HCC * ZD);
  cvt(x, b_x, S);

  // ---- GINE local branch ----
  fill(f_aggr, 0.f, S);
  k_gine_edges<<<dim3(E), dim3(256), 0, stream>>>(x, src, dst, ew, gew, geb, f_aggr, E);
  k_add<<<dim3((unsigned)((S + 255) / 256)), dim3(256), 0, stream>>>(f_aggr, f_aggr, x, S); // tmp = x+aggr
  cvt(f_aggr, b_tmp, S);
  gemm(b_tmp, 0, 0, b_gw1, 0, 0, gb1, 0, 0, f_h1, 0, 0, 1, SEGC, SEGC, SEGC,
       Nn, SEGC, SEGC, 1, 1.f, 1, 1);                                   // h1 = relu(tmp@W1^T+b1)
  cvt(f_h1, b_h1, S);
  gemm(b_h1, 0, 0, b_gw2, 0, 0, gb2, 0, 0, f_h2, 0, 0, 1, SEGC, SEGC, SEGC,
       Nn, SEGC, SEGC, 1, 1.f, 0, 1);                                   // h = h1@W2^T+b2
  k_add_scale<<<dim3((unsigned)((S + 255) / 256)), dim3(256), 0, stream>>>(f_h2, f_h2, x, BNSC, S);

  // ---- MHA global branch ----
  gemm(b_x, 0, 0, b_aiw, 0, 0, aib, 0, 0, f_qkv, 0, 0, 1, SEGC, SEGC, 3 * SEGC,
       Nn, 3 * SEGC, SEGC, 1, 1.f, 0, 1);                               // qkv
  cvt(f_qkv, b_qkv, 3 * S);
  {
    const float inv_sqrt_dh = 0.08703882797784892f;  // 1/sqrt(132)
    long long sg = (long long)SEGC * 3 * SEGC;       // per-graph stride in qkv
    gemm(b_qkv, sg, KTOP, b_qkv + SEGC, sg, KTOP, nullptr, 0, 0,
         f_scores, 2LL * SEGC * SEGC, (long long)SEGC * SEGC,
         2, 3 * SEGC, 3 * SEGC, SEGC,
         SEGC, SEGC, KTOP, 1, inv_sqrt_dh, 0, Bn * 2);                  // q k^T / sqrt(dh)
    k_softmax_rows<<<dim3(Bn * 2 * SEGC), dim3(256), 0, stream>>>(f_scores, f_scores,
                                                                  Bn * 2 * SEGC, SEGC);
    cvt(f_scores, b_sc, (long long)Bn * 2 * SEGC * SEGC);
    gemm(b_sc, 2LL * SEGC * SEGC, (long long)SEGC * SEGC,
         b_qkv + 2 * SEGC, sg, KTOP, nullptr, 0, 0,
         f_ao, (long long)SEGC * SEGC, KTOP,
         2, SEGC, 3 * SEGC, SEGC,
         SEGC, KTOP, SEGC, 0, 1.f, 0, Bn * 2);                          // att @ v
  }
  cvt(f_ao, b_ao, S);
  gemm(b_ao, 0, 0, b_aow, 0, 0, aob, 0, 0, f_oproj, 0, 0, 1, SEGC, SEGC, SEGC,
       Nn, SEGC, SEGC, 1, 1.f, 0, 1);                                   // o proj
  k_combine<<<dim3((unsigned)((S + 255) / 256)), dim3(256), 0, stream>>>(f_out, f_h2, f_oproj, x, BNSC, S);

  // ---- MLP + residual ----
  cvt(f_out, b_out, S);
  gemm(b_out, 0, 0, b_m1w, 0, 0, m1b, 0, 0, f_t1, 0, 0, 1, SEGC, SEGC, 2 * SEGC,
       Nn, 2 * SEGC, SEGC, 1, 1.f, 1, 1);
  cvt(f_t1, b_t1, 2 * S);
  gemm(b_t1, 0, 0, b_m2w, 0, 0, m2b, 0, 0, f_t2, 0, 0, 1, 2 * SEGC, 2 * SEGC, SEGC,
       Nn, SEGC, 2 * SEGC, 1, 1.f, 0, 1);
  k_add_relu_scale<<<dim3((unsigned)((S + 255) / 256)), dim3(256), 0, stream>>>(f_x0, f_out, f_t2, BNSC, S);
  cvt(f_x0, b_x0, S);

  // ---- two GAT branches ----
  gemm(b_x0, 0, 0, b_g1w, 0, 0, nullptr, 0, 0, f_xp1, 0, 0, 1, SEGC, SEGC, HIDC,
       Nn, HIDC, SEGC, 1, 1.f, 0, 1);
  gemm(b_x0, 0, 0, b_g2w, 0, 0, nullptr, 0, 0, f_xp2, 0, 0, 1, SEGC, SEGC, HIDC,
       Nn, HIDC, SEGC, 1, 1.f, 0, 1);
  {
    unsigned gn = (Nn + 255) / 256, ge = (E + 255) / 256;
    k_gat_scores<<<gn, 256, 0, stream>>>(f_xp1, g1as, g1ad, f_as1, f_ad1, Nn);
    k_gat_scores<<<gn, 256, 0, stream>>>(f_xp2, g2as, g2ad, f_as2, f_ad2, Nn);
    fill(f_m1, -3.0e38f, Nn); fill(f_m2, -3.0e38f, Nn);
    fill(f_dn1, 0.f, Nn);     fill(f_dn2, 0.f, Nn);
    fill(f_go1, 0.f, (long long)Nn * HIDC); fill(f_go2, 0.f, (long long)Nn * HIDC);
    k_gat_max<<<ge, 256, 0, stream>>>(src, dst, E, f_as1, f_ad1, f_m1);
    k_gat_max<<<gn, 256, 0, stream>>>(nullptr, nullptr, Nn, f_as1, f_ad1, f_m1);
    k_gat_max<<<ge, 256, 0, stream>>>(src, dst, E, f_as2, f_ad2, f_m2);
    k_gat_max<<<gn, 256, 0, stream>>>(nullptr, nullptr, Nn, f_as2, f_ad2, f_m2);
    k_gat_den<<<ge, 256, 0, stream>>>(src, dst, E, f_as1, f_ad1, f_m1, f_dn1);
    k_gat_den<<<gn, 256, 0, stream>>>(nullptr, nullptr, Nn, f_as1, f_ad1, f_m1, f_dn1);
    k_gat_den<<<ge, 256, 0, stream>>>(src, dst, E, f_as2, f_ad2, f_m2, f_dn2);
    k_gat_den<<<gn, 256, 0, stream>>>(nullptr, nullptr, Nn, f_as2, f_ad2, f_m2, f_dn2);
    k_gat_aggr<<<dim3(E), dim3(HIDC), 0, stream>>>(src, dst, E, f_as1, f_ad1, f_m1, f_dn1, f_xp1, f_go1);
    k_gat_aggr<<<dim3(Nn), dim3(HIDC), 0, stream>>>(nullptr, nullptr, Nn, f_as1, f_ad1, f_m1, f_dn1, f_xp1, f_go1);
    k_gat_aggr<<<dim3(E), dim3(HIDC), 0, stream>>>(src, dst, E, f_as2, f_ad2, f_m2, f_dn2, f_xp2, f_go2);
    k_gat_aggr<<<dim3(Nn), dim3(HIDC), 0, stream>>>(nullptr, nullptr, Nn, f_as2, f_ad2, f_m2, f_dn2, f_xp2, f_go2);
    long long nh = (long long)Nn * HIDC;
    k_bias_relu<<<dim3((unsigned)((nh + 255) / 256)), 256, 0, stream>>>(f_go1, f_go1, g1b, HIDC, nh); // x11
    k_bias_relu<<<dim3((unsigned)((nh + 255) / 256)), 256, 0, stream>>>(f_go2, f_go2, g2b, HIDC, nh); // x22
  }
  float* f_x11 = f_go1; float* f_x22 = f_go2;

  // ---- node attention ----
  k_conv4<<<dim3((Nn + 255) / 256), 256, 0, stream>>>(f_x11, f_x22, c4w, c4b, f_a, Nn);
  cvt(f_a, b_a, Nn);
  gemm(b_a, 0, 0, b_l4w, 0, 0, l4b, 0, 0, f_t4, 0, 0, 1, SEGC, SEGC, 66,
       Bn, 66, SEGC, 1, 1.f, 1, 1);
  cvt(f_t4, b_t4, (long long)Bn * 66);
  gemm(b_t4, 0, 0, b_l5w, 0, 0, l5b, 0, 0, f_attn, 0, 0, 1, 66, 66, SEGC,
       Bn, SEGC, 66, 1, 1.f, 0, 1);
  k_softmax_rows<<<dim3(Bn), 256, 0, stream>>>(f_attn, f_attn, Bn, SEGC);
  {
    long long nh = (long long)Nn * HIDC;
    k_allscale<<<dim3((unsigned)((nh + 255) / 256)), 256, 0, stream>>>(f_attn, f_x11, f_x22, f_all1, f_all2, nh);
  }

  // ---- TopK pooling, z assembly, outputs perm/sc ----
  k_wn<<<dim3(8), dim3(HIDC), 0, stream>>>(pw, f_wn);
  k_topk<<<dim3(Bn, 8), dim3(SEGC), 0, stream>>>(f_x11, f_x22, f_wn, f_vals, i_idx, dout, Bn);
  {
    long long np = 8LL * Bn * KTOP * HIDC;
    long long nc = 8LL * Bn * SEGC * HIDC;
    k_zpool<<<dim3((unsigned)((np + 255) / 256)), 256, 0, stream>>>(f_x11, f_x22, f_vals, i_idx, b_z, Bn);
    k_zcat<<<dim3((unsigned)((nc + 255) / 256)), 256, 0, stream>>>(f_all1, f_all2, b_z, Bn);
  }

  // ---- 8-head big GEMM + output ----
  gemm(b_z, (long long)Bn * ZD, 0, b_lbw, (long long)HCC * ZD, 0,
       lbb, HCC, 0, f_y, (long long)Bn * HCC, 0,
       1, ZD, ZD, HCC, Bn, HCC, ZD, 1, 1.f, 1, 8);
  k_yo<<<dim3(Bn, 8), dim3(HCC), 0, stream>>>(f_y, lsw, lsb, dout, Bn);
}
